// MultiHeadSelfAttention_40802189312151
// MI455X (gfx1250) — compile-verified, matrix operand order checked by reading, not run on hardware
//
#include <hip/hip_runtime.h>
#include <hip/hip_bf16.h>

// ---------- types ----------
typedef __attribute__((ext_vector_type(16))) __bf16        v16bf;
typedef __attribute__((ext_vector_type(8)))  float         v8f;
typedef __attribute__((ext_vector_type(8)))  unsigned int  v8u;
typedef __attribute__((ext_vector_type(4)))  unsigned int  v4u;

#define NEG_INF (-__builtin_inff())

// problem constants
#define BB   4
#define TT   2048
#define CC   1024
#define HH   16
#define DD   64
#define C3   3072
#define MROWS (BB * TT)          // 8192

// ---------- helpers ----------
__device__ __forceinline__ unsigned short f2bf(float f) {
  unsigned u = __float_as_uint(f);
  unsigned r = u + 0x7fffu + ((u >> 16) & 1u);   // round to nearest even
  return (unsigned short)(r >> 16);
}

// K-offset of VGPR-pair p for the 16-bit A/B fragment layout (16x32, wave32)
__device__ __forceinline__ int kpair(int p, int khalf) {
  return ((p >> 2) << 4) + (khalf << 3) + ((p & 3) << 1);
}

// CDNA5 async memory->LDS copy (ASYNCcnt-tracked, no VGPR transit).
// LDS offset = low 32 bits of the generic shared-memory pointer
__device__ __forceinline__ void async_b128(const void* lds_dst, const void* gsrc) {
  unsigned ldsoff = (unsigned)(uintptr_t)lds_dst;
  unsigned long long ga = (unsigned long long)(uintptr_t)gsrc;
  asm volatile("global_load_async_to_lds_b128 %0, %1, off"
               :: "v"(ldsoff), "v"(ga) : "memory");
}
__device__ __forceinline__ void wait_async0() {
  asm volatile("s_wait_asynccnt 0x0" ::: "memory");
}

// ---------- kernel 0: f32 -> bf16 conversion ----------
__global__ void cvt_bf16(const float* __restrict__ in,
                         unsigned short* __restrict__ out, int n) {
  int i = (blockIdx.x * 256 + threadIdx.x) * 4;
  if (i + 3 < n) {
    float4 v = *(const float4*)(in + i);
    unsigned lo = (unsigned)f2bf(v.x) | ((unsigned)f2bf(v.y) << 16);
    unsigned hi = (unsigned)f2bf(v.z) | ((unsigned)f2bf(v.w) << 16);
    uint2 pk; pk.x = lo; pk.y = hi;
    *(uint2*)(out + i) = pk;
  }
}

// ---------- kernel 1/3: GEMM  C[M,N] = A[M,K] * B[N,K]^T + bias ----------
// 128x128 block tile, K-step 32, 8 waves: 2(M) x 4(N), per-wave 64x32.
// Double-buffered LDS, tiles filled with async global->LDS copies that
// overlap the previous tile's WMMAs.
__global__ __launch_bounds__(256) void gemm_bf16(
    const unsigned short* __restrict__ A,   // [M x K] bf16 row-major
    const unsigned short* __restrict__ Bm,  // [N x K] bf16 row-major
    const float* __restrict__ bias,         // [N]
    unsigned short* __restrict__ Obf,       // bf16 out (or null)
    float* __restrict__ Of,                 // f32 out  (or null)
    int K, int ldo) {
  __shared__ unsigned short ldsA[2][128 * 40];
  __shared__ unsigned short ldsB[2][128 * 40];

  const int tid = threadIdx.x;
  const int lane = tid & 31, wave = tid >> 5;
  const int khalf = lane >> 4, lr = lane & 15;
  const int wm = (wave & 1) * 64, wn = (wave >> 1) * 32;
  const int m0 = blockIdx.y * 128, n0 = blockIdx.x * 128;

  const int r0 = tid >> 2, s0 = (tid & 3) * 8;       // chunk 0: rows 0..63
  const int r1 = r0 + 64;                            // chunk 1: rows 64..127

  v8f acc[4][2] = {};
  const int nk = K / 32;

  // prologue: issue tile 0
  {
    const unsigned short* a0 = A + (size_t)(m0 + r0) * K + s0;
    const unsigned short* a1 = A + (size_t)(m0 + r1) * K + s0;
    const unsigned short* b0 = Bm + (size_t)(n0 + r0) * K + s0;
    const unsigned short* b1 = Bm + (size_t)(n0 + r1) * K + s0;
    async_b128(&ldsA[0][r0 * 40 + s0], a0);
    async_b128(&ldsA[0][r1 * 40 + s0], a1);
    async_b128(&ldsB[0][r0 * 40 + s0], b0);
    async_b128(&ldsB[0][r1 * 40 + s0], b1);
  }

  for (int t = 0; t < nk; ++t) {
    const int cur = t & 1;
    wait_async0();
    __syncthreads();
    if (t + 1 < nk) {  // prefetch next tile into the other buffer
      int kn = (t + 1) * 32;
      async_b128(&ldsA[cur ^ 1][r0 * 40 + s0], A + (size_t)(m0 + r0) * K + kn + s0);
      async_b128(&ldsA[cur ^ 1][r1 * 40 + s0], A + (size_t)(m0 + r1) * K + kn + s0);
      async_b128(&ldsB[cur ^ 1][r0 * 40 + s0], Bm + (size_t)(n0 + r0) * K + kn + s0);
      async_b128(&ldsB[cur ^ 1][r1 * 40 + s0], Bm + (size_t)(n0 + r1) * K + kn + s0);
    }

    const unsigned short* la = ldsA[cur];
    const unsigned short* lb = ldsB[cur];
    v16bf af[4], bfr[2];
#pragma unroll
    for (int mi = 0; mi < 4; ++mi) {
      int row = wm + mi * 16 + lr;
      v8u tt;
#pragma unroll
      for (int p = 0; p < 8; ++p)
        tt[p] = *(const unsigned int*)(la + row * 40 + kpair(p, khalf));
      af[mi] = __builtin_bit_cast(v16bf, tt);
    }
#pragma unroll
    for (int ni = 0; ni < 2; ++ni) {
      int row = wn + ni * 16 + lr;
      v8u tt;
#pragma unroll
      for (int p = 0; p < 8; ++p)
        tt[p] = *(const unsigned int*)(lb + row * 40 + kpair(p, khalf));
      bfr[ni] = __builtin_bit_cast(v16bf, tt);
    }
#pragma unroll
    for (int mi = 0; mi < 4; ++mi)
#pragma unroll
      for (int ni = 0; ni < 2; ++ni)
        acc[mi][ni] = __builtin_amdgcn_wmma_f32_16x16x32_bf16(
            false, af[mi], false, bfr[ni], (short)0, acc[mi][ni], false, false);
  }

#pragma unroll
  for (int mi = 0; mi < 4; ++mi)
#pragma unroll
    for (int ni = 0; ni < 2; ++ni) {
      int n = n0 + wn + ni * 16 + lr;
      float bv = bias ? bias[n] : 0.f;
#pragma unroll
      for (int r = 0; r < 8; ++r) {
        int m = m0 + wm + mi * 16 + r + 8 * khalf;
        float v = acc[mi][ni][r] + bv;
        if (Obf) Obf[(size_t)m * ldo + n] = f2bf(v);
        else     Of[(size_t)m * ldo + n] = v;
      }
    }
}

// ---------- kernel 2: fused causal attention (per (b,h,128-query block)) ----
// pass1: online softmax stats via QK^T WMMA (double-buffered async K tiles);
// pass2: recompute scores, write normalized attn weights once, accumulate P*V.
__global__ __launch_bounds__(256) void attn_fused(
    const unsigned short* __restrict__ qkv,  // [B*T, 3072] bf16 (Q|K|V)
    float* __restrict__ attn,                // [B*H, T, T] f32
    unsigned short* __restrict__ ao) {       // [B*T, 1024] bf16
  __shared__ unsigned short ldsQ[128 * 72];
  __shared__ unsigned short ldsK[2][32 * 72];
  __shared__ unsigned short ldsVt[64 * 40];  // V transposed: [d][key]
  __shared__ unsigned short ldsP[8 * 16 * 34];

  const int tid = threadIdx.x;
  const int lane = tid & 31, wave = tid >> 5;
  const int khalf = lane >> 4, lr = lane & 15;

  const int qblk = blockIdx.x & 15;
  const int bh = blockIdx.x >> 4;
  const int b = bh >> 4, h = bh & 15;
  const int q0 = qblk * 128;
  const int nkt = q0 / 32 + 4;  // causal: #active 32-key tiles

  const int krow = tid >> 3, kseg = (tid & 7) * 8;   // per-thread K/V chunk

  // ---- async-load Q tile (128 x 64) + first K tile ----
#pragma unroll
  for (int i = 0; i < 4; ++i) {
    int c = tid + i * 256;
    int row = c >> 3, seg = (c & 7) * 8;
    async_b128(&ldsQ[row * 72 + seg],
               qkv + (size_t)(b * TT + q0 + row) * C3 + h * 64 + seg);
  }
  async_b128(&ldsK[0][krow * 72 + kseg],
             qkv + (size_t)(b * TT + krow) * C3 + CC + h * 64 + kseg);
  wait_async0();
  __syncthreads();

  // hoisted A fragments of Q (contraction over d: two K=32 steps)
  v16bf aq[2];
  {
    int row = wave * 16 + lr;
#pragma unroll
    for (int cc = 0; cc < 2; ++cc) {
      v8u t;
#pragma unroll
      for (int p = 0; p < 8; ++p)
        t[p] = *(const unsigned int*)(ldsQ + row * 72 + cc * 32 + kpair(p, khalf));
      aq[cc] = __builtin_bit_cast(v16bf, t);
    }
  }

  float m_run[8], l_run[8];
#pragma unroll
  for (int r = 0; r < 8; ++r) { m_run[r] = NEG_INF; l_run[r] = 0.f; }

  // ---- pass 1: softmax stats (double-buffered async K pipeline) ----
  for (int kt = 0; kt < nkt; ++kt) {
    const int cur = kt & 1;
    wait_async0();
    __syncthreads();
    if (kt + 1 < nkt)
      async_b128(&ldsK[cur ^ 1][krow * 72 + kseg],
                 qkv + (size_t)(b * TT + (kt + 1) * 32 + krow) * C3 + CC + h * 64 + kseg);

    const unsigned short* lk = ldsK[cur];
    v8f sc[2] = {};
#pragma unroll
    for (int j = 0; j < 2; ++j) {
      int row = j * 16 + lr;
#pragma unroll
      for (int cc = 0; cc < 2; ++cc) {
        v8u t;
#pragma unroll
        for (int p = 0; p < 8; ++p)
          t[p] = *(const unsigned int*)(lk + row * 72 + cc * 32 + kpair(p, khalf));
        v16bf bk = __builtin_bit_cast(v16bf, t);
        sc[j] = __builtin_amdgcn_wmma_f32_16x16x32_bf16(
            false, aq[cc], false, bk, (short)0, sc[j], false, false);
      }
    }
#pragma unroll
    for (int r = 0; r < 8; ++r) {
      int q = q0 + wave * 16 + r + 8 * khalf;
      float v0 = sc[0][r] * 0.125f;
      float v1 = sc[1][r] * 0.125f;
      int k0i = kt * 32 + lr, k1i = kt * 32 + 16 + lr;
      if (k0i > q) v0 = NEG_INF;
      if (k1i > q) v1 = NEG_INF;
      float tm = fmaxf(v0, v1);
#pragma unroll
      for (int off = 8; off > 0; off >>= 1) tm = fmaxf(tm, __shfl_xor(tm, off, 32));
      float mn = fmaxf(m_run[r], tm);
      float ps = __expf(v0 - mn) + __expf(v1 - mn);
#pragma unroll
      for (int off = 8; off > 0; off >>= 1) ps += __shfl_xor(ps, off, 32);
      l_run[r] = l_run[r] * __expf(m_run[r] - mn) + ps;
      m_run[r] = mn;
    }
  }

  float inv_l[8];
#pragma unroll
  for (int r = 0; r < 8; ++r) inv_l[r] = 1.0f / l_run[r];

  // ---- pass 2: recompute scores, write P, accumulate P*V ----
  float* arow = attn + (size_t)bh * TT * TT;
  v8f o[4] = {};

  for (int kt = 0; kt < 64; ++kt) {
    if (kt < nkt) {
      __syncthreads();  // previous iteration's LDS reads are done
      // K tile: async straight into LDS
      async_b128(&ldsK[0][krow * 72 + kseg],
                 qkv + (size_t)(b * TT + kt * 32 + krow) * C3 + CC + h * 64 + kseg);
      // V tile: load + transpose into [d][key]
      {
        v4u vv = *(const v4u*)(qkv + (size_t)(b * TT + kt * 32 + krow) * C3 + 2 * CC + h * 64 + kseg);
        unsigned short e[8];
        *(v4u*)&e[0] = vv;
#pragma unroll
        for (int i = 0; i < 8; ++i) ldsVt[(kseg + i) * 40 + krow] = e[i];
      }
      wait_async0();
      __syncthreads();

      v8f sc[2] = {};
#pragma unroll
      for (int j = 0; j < 2; ++j) {
        int row = j * 16 + lr;
#pragma unroll
        for (int cc = 0; cc < 2; ++cc) {
          v8u t;
#pragma unroll
          for (int p = 0; p < 8; ++p)
            t[p] = *(const unsigned int*)(ldsK[0] + row * 72 + cc * 32 + kpair(p, khalf));
          v16bf bk = __builtin_bit_cast(v16bf, t);
          sc[j] = __builtin_amdgcn_wmma_f32_16x16x32_bf16(
              false, aq[cc], false, bk, (short)0, sc[j], false, false);
        }
      }
      // normalize, write attn weights, stash bf16 P into per-wave LDS
#pragma unroll
      for (int j = 0; j < 2; ++j)
#pragma unroll
        for (int r = 0; r < 8; ++r) {
          int m = r + 8 * khalf;
          int q = q0 + wave * 16 + m;
          int ki = kt * 32 + j * 16 + lr;
          float v = sc[j][r] * 0.125f;
          if (ki > q) v = NEG_INF;
          float pp = __expf(v - m_run[r]) * inv_l[r];
          arow[(size_t)q * TT + ki] = pp;
          ldsP[wave * 544 + m * 34 + j * 16 + lr] = f2bf(pp);
        }
      // re-gather P as A fragment (same wave wrote it; LDS is in-order/wave)
      v8u pu;
#pragma unroll
      for (int p = 0; p < 8; ++p)
        pu[p] = *(const unsigned int*)(ldsP + wave * 544 + lr * 34 + kpair(p, khalf));
      v16bf ap = __builtin_bit_cast(v16bf, pu);
#pragma unroll
      for (int nt = 0; nt < 4; ++nt) {
        int dcol = nt * 16 + lr;
        v8u t;
#pragma unroll
        for (int p = 0; p < 8; ++p)
          t[p] = *(const unsigned int*)(ldsVt + dcol * 40 + kpair(p, khalf));
        v16bf bv = __builtin_bit_cast(v16bf, t);
        o[nt] = __builtin_amdgcn_wmma_f32_16x16x32_bf16(
            false, ap, false, bv, (short)0, o[nt], false, false);
      }
    } else {
      // fully-masked tile: attention weights are exactly zero
#pragma unroll
      for (int j = 0; j < 2; ++j)
#pragma unroll
        for (int r = 0; r < 8; ++r) {
          int q = q0 + wave * 16 + r + 8 * khalf;
          arow[(size_t)q * TT + kt * 32 + j * 16 + lr] = 0.f;
        }
    }
  }

  // store attention output (bf16) as [B*T, C] with head h at cols h*64..
#pragma unroll
  for (int nt = 0; nt < 4; ++nt) {
    int d = nt * 16 + lr;
#pragma unroll
    for (int r = 0; r < 8; ++r) {
      int q = q0 + wave * 16 + r + 8 * khalf;
      ao[(size_t)(b * TT + q) * CC + h * 64 + d] = f2bf(o[nt][r]);
    }
  }
}

// ---------- launch ----------
extern "C" void kernel_launch(void* const* d_in, const int* in_sizes, int n_in,
                              void* d_out, int out_size, void* d_ws, size_t ws_size,
                              hipStream_t stream) {
  const float* x      = (const float*)d_in[0];   // [B,T,C]
  const float* qkv_w  = (const float*)d_in[1];   // [3C,C]
  const float* qkv_b  = (const float*)d_in[2];   // [3C]
  const float* out_w  = (const float*)d_in[3];   // [C,C]
  const float* out_b  = (const float*)d_in[4];   // [C]

  char* ws = (char*)d_ws;
  size_t off = 0;
  unsigned short* x_bf    = (unsigned short*)(ws + off); off += (size_t)MROWS * CC * 2;      // 16 MB
  unsigned short* qkvw_bf = (unsigned short*)(ws + off); off += (size_t)C3 * CC * 2;         //  6 MB
  unsigned short* outw_bf = (unsigned short*)(ws + off); off += (size_t)CC * CC * 2;         //  2 MB
  unsigned short* qkv_bf  = (unsigned short*)(ws + off); off += (size_t)MROWS * C3 * 2;      // 48 MB
  unsigned short* ao_bf   = (unsigned short*)(ws + off); off += (size_t)MROWS * CC * 2;      // 16 MB

  float* out_f  = (float*)d_out;                                   // [B,T,C]
  float* attn_f = (float*)d_out + (size_t)BB * TT * CC;            // [B,H,T,T]

  // 0) f32 -> bf16 operand conversion
  {
    int n;
    n = MROWS * CC; cvt_bf16<<<(n / 4 + 255) / 256, 256, 0, stream>>>(x, x_bf, n);
    n = C3 * CC;    cvt_bf16<<<(n / 4 + 255) / 256, 256, 0, stream>>>(qkv_w, qkvw_bf, n);
    n = CC * CC;    cvt_bf16<<<(n / 4 + 255) / 256, 256, 0, stream>>>(out_w, outw_bf, n);
  }

  // 1) QKV projection: [8192,3072] = x @ qkv_w^T + qkv_b  -> bf16
  gemm_bf16<<<dim3(C3 / 128, MROWS / 128), 256, 0, stream>>>(
      x_bf, qkvw_bf, qkv_b, qkv_bf, nullptr, CC, C3);

  // 2) fused causal attention: writes attn weights (f32) + ao (bf16)
  attn_fused<<<BB * HH * (TT / 128), 256, 0, stream>>>(qkv_bf, attn_f, ao_bf);

  // 3) output projection: [8192,1024] = ao @ out_w^T + out_b -> f32 d_out
  gemm_bf16<<<dim3(CC / 128, MROWS / 128), 256, 0, stream>>>(
      ao_bf, outw_bf, out_b, nullptr, out_f, CC, CC);

  (void)in_sizes; (void)n_in; (void)out_size; (void)ws_size;
}
